// BufferedCIFModule_85383949844808
// MI455X (gfx1250) — matrix operand order repaired and checked by reading, not compiled, not run on hardware
//
#include <hip/hip_runtime.h>

#define B_ 4
#define T_ 16384
#define H_ 512
#define C_ 256
#define K_ 5
#define KK_ (H_ * K_)   // 2560 (flattened conv K-dim: kk = ktap*H + h)
#define THR_ 1.0f
#define LN_EPS_ 1e-5f

#define TT_ 32                 // timesteps per conv block
#define ROWS_ (TT_ + 4)        // 36 staged input rows (pad=2 each side)
#define CHUNKS_ (ROWS_ * H_ * 4 / 16)   // 4608 16-byte staging chunks
#define XS_BYTES_ (ROWS_ * H_ * 2)      // 36864 B f16 window (fits DS imm field)

typedef __attribute__((ext_vector_type(16))) _Float16 v16h;
typedef __attribute__((ext_vector_type(8)))  _Float16 v8h;
typedef __attribute__((ext_vector_type(4)))  _Float16 v4h;
typedef __attribute__((ext_vector_type(8)))  float    v8f;

// ---------------------------------------------------------------------------
// Prep: transpose conv weights [C,H,K] (OIH) -> f16 [C][KK] with kk = k*H + h
// so a B-fragment lane reads 16 contiguous K values for its column c.
// ---------------------------------------------------------------------------
__global__ void prep_w_kernel(const float* __restrict__ conv_w,
                              _Float16* __restrict__ wT) {
  int idx = blockIdx.x * blockDim.x + threadIdx.x;   // over C_*KK_
  if (idx >= C_ * KK_) return;
  int c = idx / KK_, kk = idx % KK_;
  int ktap = kk / H_, h = kk % H_;
  wT[idx] = (_Float16)conv_w[(c * H_ + h) * K_ + ktap];
}

// ---------------------------------------------------------------------------
// Fused Conv1d(H->C,k=5,pad=2) + LayerNorm(C) + ReLU + Linear(C->1) + sigmoid.
// Block = 512 threads = 16 waves; block handles 32 timesteps.
// Stage 1: async global->LDS copy of the 36-row f32 input window.
// Stage 2: one cooperative f32->f16 conversion pass in LDS.
// Stage 3: wave w computes c-tile [16w,16w+16) for two 16-t tiles via
//          v_wmma_f32_16x16x32_f16 (160 WMMA per wave); A from ds_load_b128
//          with immediate offsets (f16 window at LDS offset 0), B from
//          L2-resident transposed weights.
// Stage 4: LayerNorm + ReLU + Linear + sigmoid through LDS (aliases staging).
// ---------------------------------------------------------------------------
__global__ void __launch_bounds__(512)
conv_ln_alpha_kernel(const float* __restrict__ x,        // [B,T,H] f32
                     const _Float16* __restrict__ wT,    // [C][KK] f16
                     const float* __restrict__ conv_b,
                     const float* __restrict__ ln_w,
                     const float* __restrict__ ln_b,
                     const float* __restrict__ lin_w,
                     const float* __restrict__ lin_b,
                     float* __restrict__ alpha) {        // [B,T]
  // smem layout: [0, 36864)        f16 converted window (DS imm-offset reach)
  //              [36864, 110592)   f32 staging, later reused as y_lds
  __shared__ __align__(16) char smem[XS_BYTES_ + ROWS_ * H_ * 4];
  _Float16* xs    = (_Float16*)smem;
  float*    xs32  = (float*)(smem + XS_BYTES_);
  float*    y_lds = (float*)(smem + XS_BYTES_);         // 32*257*4 = 32896 B

  const int tid  = threadIdx.x;
  const int lane = tid & 31;
  const int wave = tid >> 5;          // 0..15 -> c tile index
  const int c0   = wave * 16;
  const int bt0  = blockIdx.x * TT_;  // T_ % TT_ == 0, blocks never cross batch
  const int b    = bt0 / T_;
  const int t0   = bt0 % T_;
  const int m    = lane & 15;
  const int hiL  = lane >> 4;
  const int n    = lane & 15;

  // ---- Stage 1: async global->LDS copy (zero-fill out-of-range pad rows) ----
  const float* xbat = x + (size_t)b * T_ * H_;
#pragma unroll
  for (int k = 0; k < CHUNKS_ / 512; ++k) {     // 9 chunks of 16 B per thread
    int ci   = tid + k * 512;                   // 128 chunks per row
    int r    = ci >> 7;
    int trow = t0 + r - 2;
    unsigned int ldsoff =
        (unsigned int)(size_t)((char*)xs32 + (size_t)(ci * 16));
    if (trow >= 0 && trow < T_) {
      unsigned long long ga = (unsigned long long)(const void*)
          (xbat + (size_t)trow * H_ + (size_t)((ci & 127) * 4));
      asm volatile("global_load_async_to_lds_b128 %0, %1, off"
                   :: "v"(ldsoff), "v"(ga) : "memory");
    } else {
      float4 z = make_float4(0.f, 0.f, 0.f, 0.f);
      *(float4*)((char*)xs32 + (size_t)(ci * 16)) = z;
    }
  }
  asm volatile("s_wait_asynccnt 0x0" ::: "memory");
  __syncthreads();

  // ---- Stage 2: convert staged f32 -> f16 (36 floats per thread) ----
#pragma unroll
  for (int k = 0; k < CHUNKS_ / 512; ++k) {
    int ci = tid + k * 512;                     // float4 index, 4608 total
    float4 f = ((const float4*)xs32)[ci];
    v4h h;
    h[0] = (_Float16)f.x; h[1] = (_Float16)f.y;
    h[2] = (_Float16)f.z; h[3] = (_Float16)f.w;
    ((v4h*)xs)[ci] = h;
  }
  __syncthreads();

  // ---- Stage 3: GEMM, two 16-row t-tiles per wave ----
  v8f acc0 = {}, acc1 = {};
  const _Float16* wcol = wT + (size_t)(c0 + n) * KK_;
  // Per-lane A base pointers; all per-iteration offsets are compile-time
  // immediates that fit the 16-bit DS offset field (max 35840+1984 < 65536).
  const _Float16* abase0 = xs + m * H_ + hiL * 16;
  const _Float16* abase1 = xs + (m + 16) * H_ + hiL * 16;
#pragma unroll
  for (int ktap = 0; ktap < K_; ++ktap) {
    const _Float16* wtap = wcol + ktap * H_;
#pragma unroll
    for (int hq = 0; hq < 16; ++hq) {
      const int h0 = hq * 32;
      const int aoff = ktap * H_ + h0;
      // B fragment 32x16 f16: lane holds 16 contiguous K (hi*16..+15), col n
      const v16h bfrag = *(const v16h*)(wtap + h0 + hiL * 16);
      // A fragments 16x32 f16: lane K-locals hi*16..+15 (contiguous)
      v8h a0l = *(const v8h*)(abase0 + aoff);
      v8h a0h = *(const v8h*)(abase0 + aoff + 8);
      v8h a1l = *(const v8h*)(abase1 + aoff);
      v8h a1h = *(const v8h*)(abase1 + aoff + 8);
      v16h a0 = __builtin_shufflevector(a0l, a0h,
          0,1,2,3,4,5,6,7,8,9,10,11,12,13,14,15);
      v16h a1 = __builtin_shufflevector(a1l, a1h,
          0,1,2,3,4,5,6,7,8,9,10,11,12,13,14,15);
      acc0 = __builtin_amdgcn_wmma_f32_16x16x32_f16(
          false, a0, false, bfrag, (short)0, acc0, false, false);
      acc1 = __builtin_amdgcn_wmma_f32_16x16x32_f16(
          false, a1, false, bfrag, (short)0, acc1, false, false);
    }
  }
  __syncthreads();   // staging f32 region is dead; reuse as y_lds

  // C/D layout: VGPR r, lanes<16 -> M=r, lanes>=16 -> M=r+8; col = c0+n
  const float bias = conv_b[c0 + n];
#pragma unroll
  for (int r = 0; r < 8; ++r) {
    y_lds[(r + hiL * 8) * 257 + (c0 + n)]      = acc0[r] + bias;
    y_lds[(16 + r + hiL * 8) * 257 + (c0 + n)] = acc1[r] + bias;
  }
  __syncthreads();

  // ---- Stage 4: LayerNorm + ReLU + Linear; wave w owns rows w and w+16 ----
#pragma unroll
  for (int half = 0; half < 2; ++half) {
    const int row = wave + half * 16;
    float s = 0.f, s2 = 0.f;
#pragma unroll
    for (int j = 0; j < 8; ++j) {
      float v = y_lds[row * 257 + lane + j * 32];
      s += v; s2 += v * v;
    }
#pragma unroll
    for (int o = 16; o > 0; o >>= 1) {
      s  += __shfl_xor(s,  o, 32);
      s2 += __shfl_xor(s2, o, 32);
    }
    const float mu   = s * (1.0f / C_);
    const float var  = s2 * (1.0f / C_) - mu * mu;
    const float rstd = rsqrtf(var + LN_EPS_);
    float z = 0.f;
#pragma unroll
    for (int j = 0; j < 8; ++j) {
      int c = lane + j * 32;
      float v = (y_lds[row * 257 + c] - mu) * rstd * ln_w[c] + ln_b[c];
      v = v > 0.f ? v : 0.f;
      z += v * lin_w[c];
    }
#pragma unroll
    for (int o = 16; o > 0; o >>= 1) z += __shfl_xor(z, o, 32);
    if (lane == 0)
      alpha[(size_t)b * T_ + t0 + row] =
          1.0f / (1.0f + __expf(-(z + lin_b[0])));
  }
}

// ---------------------------------------------------------------------------
// CIF phase A: scalar threshold scan per batch (registers only). Emits:
//   fires  [B,T] (0/1 float, straight to d_out)
//   c_in   [B,T] coefficient of x[t] in its own emit (a_u1 at fires)
//   w_carry[B,T] coefficient of x[t] in the NEXT segment (a_u2 at fires, a_t else)
//   prev   [B,T] index of previous fire (-1 if none)
//   alphas [B]   sum of alpha per batch (d_out)
// ---------------------------------------------------------------------------
__global__ void cif_scan_kernel(const float* __restrict__ alpha,
                                float* __restrict__ fires,
                                float* __restrict__ alphas,
                                float* __restrict__ c_in,
                                float* __restrict__ w_carry,
                                int* __restrict__ prev) {
  int b = threadIdx.x;
  if (b >= B_) return;
  float s = 0.f, asum = 0.f;
  int last = -1;
  size_t base = (size_t)b * T_;
  for (int t = 0; t < T_; ++t) {
    float a_t = alpha[base + t];
    asum += a_t;
    float s1   = s + a_t;
    bool  fire = s1 >= THR_;
    float a_u1 = 1.0f - s;
    float a_u2 = a_t - a_u1;
    fires[base + t]   = fire ? 1.0f : 0.0f;
    c_in[base + t]    = fire ? a_u1 : 0.0f;
    w_carry[base + t] = fire ? a_u2 : a_t;
    prev[base + t]    = last;
    if (fire) { last = t; s = a_u2; } else { s = s1; }
  }
  alphas[b] = asum;
}

// ---------------------------------------------------------------------------
// CIF phase B: fully parallel segmented gather. One block per (b,t); 128
// threads x float4 = 512 features. Non-fire rows are zero-filled (d_out is
// poisoned, so every element must be written).
// ---------------------------------------------------------------------------
__global__ void __launch_bounds__(128)
cif_emit_kernel(const float* __restrict__ x,
                const float* __restrict__ fires,
                const float* __restrict__ c_in,
                const float* __restrict__ w_carry,
                const int* __restrict__ prev,
                float* __restrict__ emits) {
  const int bt  = blockIdx.x;         // b*T + t
  const int b   = bt / T_;
  const int t   = bt % T_;
  const int tid = threadIdx.x;        // 0..127

  float4* orow = (float4*)(emits + (size_t)bt * H_) + tid;
  if (fires[bt] == 0.0f) {
    float4 z = make_float4(0.f, 0.f, 0.f, 0.f);
    *orow = z;
    return;
  }

  const float* xb = x + (size_t)b * T_ * H_;
  const float  ci = c_in[bt];
  float4 xt = *((const float4*)(xb + (size_t)t * H_) + tid);
  float4 acc;
  acc.x = ci * xt.x; acc.y = ci * xt.y; acc.z = ci * xt.z; acc.w = ci * xt.w;

  int start = prev[bt];
  if (start < 0) start = 0;
  for (int tau = start; tau < t; ++tau) {
    float w = w_carry[(size_t)b * T_ + tau];
    float4 xv = *((const float4*)(xb + (size_t)tau * H_) + tid);
    acc.x += w * xv.x; acc.y += w * xv.y;
    acc.z += w * xv.z; acc.w += w * xv.w;
  }
  *orow = acc;
}

// ---------------------------------------------------------------------------
extern "C" void kernel_launch(void* const* d_in, const int* in_sizes, int n_in,
                              void* d_out, int out_size, void* d_ws, size_t ws_size,
                              hipStream_t stream) {
  const float* x      = (const float*)d_in[0];  // encoder_out [B,T,H]
  const float* conv_w = (const float*)d_in[1];  // [C,H,K]
  const float* conv_b = (const float*)d_in[2];  // [C]
  const float* ln_w   = (const float*)d_in[3];  // [C]
  const float* ln_b   = (const float*)d_in[4];  // [C]
  const float* lin_w  = (const float*)d_in[5];  // [C]
  const float* lin_b  = (const float*)d_in[6];  // [1]

  float* emits  = (float*)d_out;                         // [B,T,H]
  float* fires  = emits + (size_t)B_ * T_ * H_;          // [B,T]
  float* alphas = fires + (size_t)B_ * T_;               // [B]

  // Workspace layout (~2.3 MB)
  char* ws = (char*)d_ws;
  _Float16* wT   = (_Float16*)ws;                        // C*KK f16 = 1,310,720 B
  float* alpha   = (float*)(ws + (size_t)C_ * KK_ * 2);  // B*T f32
  float* c_in    = alpha + (size_t)B_ * T_;
  float* w_carry = c_in  + (size_t)B_ * T_;
  int*   prev    = (int*)(w_carry + (size_t)B_ * T_);

  prep_w_kernel<<<(C_ * KK_ + 255) / 256, 256, 0, stream>>>(conv_w, wT);

  conv_ln_alpha_kernel<<<(B_ * T_) / TT_, 512, 0, stream>>>(
      x, wT, conv_b, ln_w, ln_b, lin_w, lin_b, alpha);

  cif_scan_kernel<<<1, 32, 0, stream>>>(alpha, fires, alphas, c_in, w_carry, prev);

  cif_emit_kernel<<<B_ * T_, 128, 0, stream>>>(x, fires, c_in, w_carry, prev, emits);
}
